// Mamba2Block_27238682591314
// MI455X (gfx1250) — matrix operand order repaired and checked by reading, not compiled, not run on hardware
//
#include <hip/hip_runtime.h>
#include <math.h>

// ---------------- problem constants ----------------
#define BATCH   2
#define SEQ     4096
#define DMODEL  1024
#define DINNER  2048
#define NHEADS  16
#define DSTATE  16
#define DCONV   4
#define CHUNK   64
#define HEADDIM 128
#define NCHUNK  (SEQ / CHUNK)          // 64
#define ROWS    (BATCH * SEQ)          // 8192

typedef __attribute__((ext_vector_type(16))) __bf16 v16bf;
typedef __attribute__((ext_vector_type(8)))  float  v8f;
typedef __attribute__((ext_vector_type(4)))  float  fvec4;

__device__ __forceinline__ unsigned short f2bf(float f) {
  unsigned u = __float_as_uint(f);
  u += 0x7fffu + ((u >> 16) & 1u);     // round-to-nearest-even
  return (unsigned short)(u >> 16);
}

// gfx1250 async global->LDS copy (probe-verified signature: int4 AS1*, int4 AS3*, imm, imm)
#if defined(__AMDGCN__) && __has_builtin(__builtin_amdgcn_global_load_async_to_lds_b128)
#define ASYNC_LDS 1
typedef int vi4 __attribute__((__vector_size__(16)));
typedef vi4 __attribute__((address_space(1))) as1_vi4;
typedef vi4 __attribute__((address_space(3))) as3_vi4;
__device__ __forceinline__ void async_cp16(const void* g, void* l) {
  __builtin_amdgcn_global_load_async_to_lds_b128((as1_vi4*)g, (as3_vi4*)l, 0, 0);
}
#else
#define ASYNC_LDS 0
#endif

// =====================================================================
// Streaming f32 -> bf16 pack.  permute=1 additionally swaps bits 3<->4 of
// the flat index (valid because K % 32 == 0): this pre-applies the WMMA
// A-fragment K-interleave so GEMM tile loads are straight 16B copies.
// =====================================================================
__global__ __launch_bounds__(256)
void pack_bf16_kernel(const float* __restrict__ in, unsigned short* __restrict__ out,
                      int permute)
{
  const int g = blockIdx.x * 256 + threadIdx.x;
  const size_t ib = (size_t)g * 8;
  fvec4 a = *(const fvec4*)(in + ib);
  fvec4 b = *(const fvec4*)(in + ib + 4);
  uint4 pk;
  pk.x = (unsigned)f2bf(a.x) | ((unsigned)f2bf(a.y) << 16);
  pk.y = (unsigned)f2bf(a.z) | ((unsigned)f2bf(a.w) << 16);
  pk.z = (unsigned)f2bf(b.x) | ((unsigned)f2bf(b.y) << 16);
  pk.w = (unsigned)f2bf(b.z) | ((unsigned)f2bf(b.w) << 16);
  int go = g;
  if (permute) go = (g & ~3) | ((g & 1) << 1) | ((g >> 1) & 1);  // swap idx bits 3,4
  *(uint4*)(out + (size_t)go * 8) = pk;
}

// =====================================================================
// bf16 WMMA GEMM:  C[M,N] = A[M,K] x B[K,N], f32 accum.
//  - A pre-permuted bf16 (K-interleaved), B plain bf16.
//  - 256 threads = 8 waves (2x4); wave tile 64x32 = 4x2 wmma tiles.
//  - BM=128, BN=128, BK=64 -> 16 v_wmma per k-block per wave.
//  - DOUBLE-BUFFERED LDS: tile i+1 streamed via async-to-LDS while tile i
//    is consumed by WMMA (one barrier + one s_wait_asynccnt per k-block).
// Epilogue splits columns: col<splitN -> C0, else -> C1.
// =====================================================================
#define BM 128
#define BN 128
#define BK 64
#define LDA 72    // u16 stride (144B: 16B-aligned, bank-spread), row len 64
#define LDB 136   // u16 stride (272B: 16B-aligned), row len 128

__global__ __launch_bounds__(256)
void gemm_bf16_wmma(const unsigned short* __restrict__ A,
                    const unsigned short* __restrict__ B,
                    float* __restrict__ C0, float* __restrict__ C1,
                    int M, int N, int K, int splitN)
{
  __shared__ __align__(16) unsigned short Al[2][BM * LDA];
  __shared__ __align__(16) unsigned short Bl[2][BK * LDB];

  const int tid  = threadIdx.x;
  const int lane = tid & 31;
  const int wave = tid >> 5;
  const int wr   = wave >> 2;          // 0..1  (64-row band)
  const int wc   = wave & 3;           // 0..3  (32-col band)
  const int m0   = blockIdx.x * BM;
  const int n0   = blockIdx.y * BN;

  const v8f vz = {};
  v8f acc[4][2];
  #pragma unroll
  for (int mi = 0; mi < 4; ++mi) { acc[mi][0] = vz; acc[mi][1] = vz; }

  // tile loaders: 32 u16 (64B) per thread per tile
  const int ar   = tid >> 1;           // 0..127
  const int aseg = (tid & 1) * 32;
  const int brr  = tid >> 2;           // 0..63
  const int bseg = (tid & 3) * 32;
  const unsigned short* agp = A + (size_t)(m0 + ar) * K + aseg;
  const unsigned short* bgp = B + (size_t)brr * N + (n0 + bseg);
  unsigned short* alp[2] = { &Al[0][ar * LDA + aseg], &Al[1][ar * LDA + aseg] };
  unsigned short* blp[2] = { &Bl[0][brr * LDB + bseg], &Bl[1][brr * LDB + bseg] };

  // ---- prologue: stage tile 0 into buffer 0 ----
  {
    const unsigned short* ag = agp;
    const unsigned short* bg = bgp;
#if ASYNC_LDS
    #pragma unroll
    for (int i = 0; i < 4; ++i) async_cp16(ag + i * 8, alp[0] + i * 8);
    #pragma unroll
    for (int i = 0; i < 4; ++i) async_cp16(bg + i * 8, blp[0] + i * 8);
#else
    #pragma unroll
    for (int i = 0; i < 4; ++i) *(uint4*)(alp[0] + i * 8) = *(const uint4*)(ag + i * 8);
    #pragma unroll
    for (int i = 0; i < 4; ++i) *(uint4*)(blp[0] + i * 8) = *(const uint4*)(bg + i * 8);
#endif
  }

  const int nk = K / BK;
  for (int it = 0; it < nk; ++it) {
    const int cur = it & 1;
#if ASYNC_LDS
    asm volatile("s_wait_asynccnt 0x0" ::: "memory");   // my tile-it copies done
#endif
    __syncthreads();   // all waves: tile-it staged, tile-(it-1) buffer free

    if (it + 1 < nk) {                 // stream tile it+1 into the spare buffer
      const int nxt = 1 - cur;
      const int k1 = (it + 1) * BK;
      const unsigned short* ag = agp + k1;
      const unsigned short* bg = bgp + (size_t)k1 * N;
#if ASYNC_LDS
      #pragma unroll
      for (int i = 0; i < 4; ++i) async_cp16(ag + i * 8, alp[nxt] + i * 8);
      #pragma unroll
      for (int i = 0; i < 4; ++i) async_cp16(bg + i * 8, blp[nxt] + i * 8);
#else
      uint4 av[4], bv[4];
      #pragma unroll
      for (int i = 0; i < 4; ++i) av[i] = *(const uint4*)(ag + i * 8);
      #pragma unroll
      for (int i = 0; i < 4; ++i) bv[i] = *(const uint4*)(bg + i * 8);
      #pragma unroll
      for (int i = 0; i < 4; ++i) *(uint4*)(alp[nxt] + i * 8) = av[i];
      #pragma unroll
      for (int i = 0; i < 4; ++i) *(uint4*)(blp[nxt] + i * 8) = bv[i];
#endif
      if (it + 2 < nk) {
        __builtin_prefetch(ag + BK, 0, 1);
        __builtin_prefetch(bg + (size_t)BK * N, 0, 1);
      }
    }

    // ---- consume tile it from buffer cur (overlaps the async stream) ----
    const unsigned short* Ab = &Al[cur][0];
    const unsigned short* Bb = &Bl[cur][0];
    #pragma unroll
    for (int ks = 0; ks < BK; ks += 32) {
      v16bf af[4], bf[2];
      const int ahalf = ks + (lane >> 4) * 16;
      #pragma unroll
      for (int mi = 0; mi < 4; ++mi)
        __builtin_memcpy(&af[mi],
            &Ab[(wr * 64 + mi * 16 + (lane & 15)) * LDA + ahalf], 32);
      #pragma unroll
      for (int ni = 0; ni < 2; ++ni)
        #pragma unroll
        for (int s = 0; s < 16; ++s)
          bf[ni][s] = __builtin_bit_cast(__bf16,
              Bb[(ks + (s & 7) + (lane >> 4) * 8 + (s >> 3) * 16) * LDB + wc * 32 + ni * 16 + (lane & 15)]);
      #pragma unroll
      for (int mi = 0; mi < 4; ++mi)
        #pragma unroll
        for (int ni = 0; ni < 2; ++ni)
          acc[mi][ni] = __builtin_amdgcn_wmma_f32_16x16x32_bf16(
              false, af[mi], false, bf[ni], (short)0, acc[mi][ni],
              false, false);
    }
  }

  // epilogue: C/D layout: vgpr r -> M = r + (lane>>4)*8, N = lane&15
  const int rbase = (lane >> 4) * 8;
  const int cl    = lane & 15;
  #pragma unroll
  for (int mi = 0; mi < 4; ++mi)
    #pragma unroll
    for (int ni = 0; ni < 2; ++ni)
      #pragma unroll
      for (int r = 0; r < 8; ++r) {
        const int row = m0 + wr * 64 + mi * 16 + rbase + r;
        const int col = n0 + wc * 32 + ni * 16 + cl;
        const float v = acc[mi][ni][r];
        if (col < splitN) C0[(size_t)row * splitN + col] = v;
        else              C1[(size_t)row * (N - splitN) + (col - splitN)] = v;
      }
}

// =====================================================================
// Depthwise causal conv (D_CONV=4) + SiLU.  One thread per (b,l,d).
// =====================================================================
__global__ __launch_bounds__(256)
void conv_silu_kernel(const float* __restrict__ xe, const float* __restrict__ conv_w,
                      const float* __restrict__ conv_b, float* __restrict__ xc)
{
  const int idx = blockIdx.x * 256 + threadIdx.x;
  const int d = idx & (DINNER - 1);
  const int l = (idx >> 11) & (SEQ - 1);
  const int b = idx >> 23;
  float acc = conv_b[d];
  #pragma unroll
  for (int j = 0; j < DCONV; ++j) {
    const int li = l - (DCONV - 1) + j;
    if (li >= 0)
      acc += conv_w[d * DCONV + j] * xe[((size_t)(b * SEQ + li)) * DINNER + d];
  }
  xc[idx] = acc / (1.f + expf(-acc));
}

// =====================================================================
// xproj: per token, 48 dots of length 2048.  8 waves x 6 outputs each.
// =====================================================================
__global__ __launch_bounds__(256)
void xproj_kernel(const float* __restrict__ xc, const float* __restrict__ Wx,
                  float* __restrict__ dtraw, float* __restrict__ Bm,
                  float* __restrict__ Cm)
{
  const int row  = blockIdx.x;
  const int lane = threadIdx.x & 31;
  const int wave = threadIdx.x >> 5;
  const float* xr = xc + (size_t)row * DINNER;
  const int obase = wave * 6;
  float acc[6] = {0.f, 0.f, 0.f, 0.f, 0.f, 0.f};
  for (int d = lane; d < DINNER; d += 32) {
    const float xv = xr[d];
    const float* w = Wx + (size_t)d * 48 + obase;
    #pragma unroll
    for (int o = 0; o < 6; ++o) acc[o] += xv * w[o];
  }
  #pragma unroll
  for (int o = 0; o < 6; ++o)
    #pragma unroll
    for (int off = 16; off >= 1; off >>= 1)
      acc[o] += __shfl_xor(acc[o], off, 32);
  if (lane == 0) {
    #pragma unroll
    for (int o = 0; o < 6; ++o) {
      const int oo = obase + o;
      const float v = acc[o];
      if (oo < 16)      dtraw[(size_t)row * 16 + oo]     = v;
      else if (oo < 32) Bm[(size_t)row * 16 + (oo - 16)] = v;
      else              Cm[(size_t)row * 16 + (oo - 32)] = v;
    }
  }
}

// =====================================================================
// Pass 1: chunk-local scan (zero carry-in).  Block = (b,h,c), thread = p.
// dt computed inline: softplus(dt_raw.Wdt + b_dt).
// =====================================================================
__global__ __launch_bounds__(128)
void scan_chunk_kernel(const float* __restrict__ xc, const float* __restrict__ dtraw,
                       const float* __restrict__ Bm, const float* __restrict__ Cm,
                       const float* __restrict__ W_dt, const float* __restrict__ b_dt,
                       const float* __restrict__ A_log, const float* __restrict__ D_skip,
                       float* __restrict__ ypart, float* __restrict__ cumdec,
                       float* __restrict__ carryS, float* __restrict__ carryD)
{
  const int blk = blockIdx.x;
  const int c = blk % NCHUNK;
  const int h = (blk / NCHUNK) % NHEADS;
  const int b = blk / (NCHUNK * NHEADS);
  const int p = threadIdx.x;
  const int d = h * HEADDIM + p;

  const float Aneg = -expf(A_log[h]);
  const float Dski = D_skip[h];
  float wdt[NHEADS];
  #pragma unroll
  for (int hh = 0; hh < NHEADS; ++hh) wdt[hh] = W_dt[hh * DINNER + d];
  const float bdt = b_dt[d];

  float st[DSTATE];
  #pragma unroll
  for (int n = 0; n < DSTATE; ++n) st[n] = 0.f;
  float cum = 0.f, cdp = 1.f;

  for (int t = 0; t < CHUNK; ++t) {
    const int l = c * CHUNK + t;
    const size_t base = (size_t)(b * SEQ + l);
    const float* dr = dtraw + base * 16;
    float dv = bdt;
    #pragma unroll
    for (int hh = 0; hh < 16; ++hh) dv += wdt[hh] * dr[hh];
    dv = (dv > 20.f) ? dv : log1pf(expf(dv));       // softplus

    const float x = xc[base * DINNER + d];
    cum += dv;
    const float dec = expf(Aneg * cum);
    cdp *= dec;
    cumdec[base * DINNER + d] = cdp;

    const float* Bp = Bm + base * 16;
    const float* Cp = Cm + base * 16;
    const float coef = dv * x;
    float y = 0.f;
    #pragma unroll
    for (int n = 0; n < DSTATE; ++n) {
      st[n] = st[n] * dec + coef * Bp[n];
      y += st[n] * Cp[n];
    }
    ypart[base * DINNER + d] = y + x * Dski;
  }
  const size_t ci = (size_t)blk;
  #pragma unroll
  for (int n = 0; n < DSTATE; ++n)
    carryS[ci * HEADDIM * DSTATE + p * DSTATE + n] = st[n];
  carryD[ci * HEADDIM + p] = cdp;
}

// =====================================================================
// Pass 2: serial prefix over the 64 chunks.  Block = (b,h), thread = p.
// =====================================================================
__global__ __launch_bounds__(128)
void chunk_prefix_kernel(const float* __restrict__ carryS,
                         const float* __restrict__ carryD,
                         float* __restrict__ stateIn)
{
  const int bh = blockIdx.x;
  const int p  = threadIdx.x;
  float st[DSTATE];
  #pragma unroll
  for (int n = 0; n < DSTATE; ++n) st[n] = 0.f;
  for (int c = 0; c < NCHUNK; ++c) {
    const size_t ci = (size_t)bh * NCHUNK + c;
    #pragma unroll
    for (int n = 0; n < DSTATE; ++n)
      stateIn[ci * HEADDIM * DSTATE + p * DSTATE + n] = st[n];
    const float dtot = carryD[ci * HEADDIM + p];
    #pragma unroll
    for (int n = 0; n < DSTATE; ++n)
      st[n] = st[n] * dtot + carryS[ci * HEADDIM * DSTATE + p * DSTATE + n];
  }
}

// =====================================================================
// Pass 3 fused with gate + LayerNorm.  Emits bf16, K-permuted, so it is
// directly the A operand of the final WMMA GEMM.
// =====================================================================
__global__ __launch_bounds__(256)
void gate_ln_kernel(const float* __restrict__ ypart, const float* __restrict__ cumdec,
                    const float* __restrict__ stateIn, const float* __restrict__ Cm,
                    const float* __restrict__ z, const float* __restrict__ ln_g,
                    const float* __restrict__ ln_b, unsigned short* __restrict__ ynbf)
{
  const int row = blockIdx.x;
  const int b = row / SEQ;
  const int l = row - b * SEQ;
  const int c = l / CHUNK;
  const int tid = threadIdx.x;

  float Cv[16];
  const float* Cp = Cm + (size_t)row * 16;
  #pragma unroll
  for (int n = 0; n < 16; ++n) Cv[n] = Cp[n];

  float vals[8];
  float s = 0.f, s2 = 0.f;
  #pragma unroll
  for (int i = 0; i < 8; ++i) {
    const int d = tid * 8 + i;
    const int h = d >> 7;
    const int p = d & 127;
    const size_t ci = (size_t)(b * NHEADS + h) * NCHUNK + c;
    const float* sp = stateIn + ci * HEADDIM * DSTATE + p * DSTATE;
    float corr = 0.f;
    #pragma unroll
    for (int n = 0; n < 16; ++n) corr += sp[n] * Cv[n];
    const float yv = ypart[(size_t)row * DINNER + d]
                   + cumdec[(size_t)row * DINNER + d] * corr;
    const float zv = z[(size_t)row * DINNER + d];
    const float g = yv * (zv / (1.f + expf(-zv)));
    vals[i] = g;
    s += g; s2 += g * g;
  }
  #pragma unroll
  for (int off = 16; off >= 1; off >>= 1) {
    s  += __shfl_xor(s,  off, 32);
    s2 += __shfl_xor(s2, off, 32);
  }
  __shared__ float red[16];
  const int lane = tid & 31, wave = tid >> 5;
  if (lane == 0) { red[wave] = s; red[8 + wave] = s2; }
  __syncthreads();
  float ts = 0.f, ts2 = 0.f;
  #pragma unroll
  for (int w = 0; w < 8; ++w) { ts += red[w]; ts2 += red[8 + w]; }
  const float mu  = ts / (float)DINNER;
  const float var = ts2 / (float)DINNER - mu * mu;
  const float inv = rsqrtf(var + 1e-5f);

  unsigned short q[8];
  #pragma unroll
  for (int i = 0; i < 8; ++i) {
    const int d = tid * 8 + i;
    q[i] = f2bf((vals[i] - mu) * inv * ln_g[d] + ln_b[d]);
  }
  // permuted group: swap bits 0,1 of the 8-element group index (= tid)
  const int gsw = (tid & ~3) | ((tid & 1) << 1) | ((tid >> 1) & 1);
  uint4 pk;
  pk.x = (unsigned)q[0] | ((unsigned)q[1] << 16);
  pk.y = (unsigned)q[2] | ((unsigned)q[3] << 16);
  pk.z = (unsigned)q[4] | ((unsigned)q[5] << 16);
  pk.w = (unsigned)q[6] | ((unsigned)q[7] << 16);
  *(uint4*)(ynbf + (size_t)row * DINNER + gsw * 8) = pk;
}

// =====================================================================
// Launch.  Workspace: ~370 MB f32 + ~63 MB bf16 in d_ws.
// =====================================================================
extern "C" void kernel_launch(void* const* d_in, const int* in_sizes, int n_in,
                              void* d_out, int out_size, void* d_ws, size_t ws_size,
                              hipStream_t stream) {
  (void)in_sizes; (void)n_in; (void)out_size; (void)ws_size;
  const float* x      = (const float*)d_in[0];
  const float* W_in   = (const float*)d_in[1];
  const float* conv_w = (const float*)d_in[2];
  const float* conv_b = (const float*)d_in[3];
  const float* W_xprj = (const float*)d_in[4];
  const float* W_dt   = (const float*)d_in[5];
  const float* b_dt   = (const float*)d_in[6];
  const float* A_log  = (const float*)d_in[7];
  const float* D_skip = (const float*)d_in[8];
  const float* W_out  = (const float*)d_in[9];
  const float* ln_g   = (const float*)d_in[10];
  const float* ln_b   = (const float*)d_in[11];

  float* ws = (float*)d_ws;
  size_t off = 0;
  const size_t NBIG = (size_t)ROWS * DINNER;            // 16,777,216
  float* xe      = ws + off; off += NBIG;
  float* z       = ws + off; off += NBIG;
  float* xc      = ws + off; off += NBIG;
  float* ypart   = ws + off; off += NBIG;
  float* cdbuf   = ws + off; off += NBIG;
  float* dtraw   = ws + off; off += (size_t)ROWS * 16;
  float* Bmb     = ws + off; off += (size_t)ROWS * 16;
  float* Cmb     = ws + off; off += (size_t)ROWS * 16;
  const size_t NCAR = (size_t)BATCH * NHEADS * NCHUNK;  // 2048
  float* carryS  = ws + off; off += NCAR * HEADDIM * DSTATE;
  float* carryD  = ws + off; off += NCAR * HEADDIM;
  float* stateIn = ws + off; off += NCAR * HEADDIM * DSTATE;

  unsigned short* u16b = (unsigned short*)(ws + off);
  size_t uoff = 0;
  unsigned short* xbf    = u16b + uoff; uoff += (size_t)ROWS * DMODEL;
  unsigned short* winbf  = u16b + uoff; uoff += (size_t)DMODEL * (2 * DINNER);
  unsigned short* woutbf = u16b + uoff; uoff += (size_t)DINNER * DMODEL;
  unsigned short* ynbf   = u16b + uoff; uoff += (size_t)ROWS * DINNER;

  // 0) one-shot bf16 packs (A-side permuted, B-side plain)
  pack_bf16_kernel<<<(ROWS * DMODEL) / 2048, 256, 0, stream>>>(x, xbf, 1);
  pack_bf16_kernel<<<(DMODEL * 2 * DINNER) / 2048, 256, 0, stream>>>(W_in, winbf, 0);
  pack_bf16_kernel<<<(DINNER * DMODEL) / 2048, 256, 0, stream>>>(W_out, woutbf, 0);

  // 1) xz = x @ W_in, split into xe | z
  dim3 g1(ROWS / BM, (2 * DINNER) / BN);                // 64 x 32
  gemm_bf16_wmma<<<g1, 256, 0, stream>>>(xbf, winbf, xe, z,
                                         ROWS, 2 * DINNER, DMODEL, DINNER);
  // 2) depthwise causal conv + SiLU
  conv_silu_kernel<<<(ROWS * DINNER) / 256, 256, 0, stream>>>(xe, conv_w, conv_b, xc);
  // 3) proj -> dt_raw | B | C
  xproj_kernel<<<ROWS, 256, 0, stream>>>(xc, W_xprj, dtraw, Bmb, Cmb);
  // 4) chunk-local scans (2048-way parallel)
  scan_chunk_kernel<<<BATCH * NHEADS * NCHUNK, 128, 0, stream>>>(
      xc, dtraw, Bmb, Cmb, W_dt, b_dt, A_log, D_skip,
      ypart, cdbuf, carryS, carryD);
  // 5) cross-chunk state prefix
  chunk_prefix_kernel<<<BATCH * NHEADS, 128, 0, stream>>>(carryS, carryD, stateIn);
  // 6) carry correction + gate + LayerNorm -> permuted bf16 A-operand
  gate_ln_kernel<<<ROWS, 256, 0, stream>>>(ypart, cdbuf, stateIn, Cmb,
                                           z, ln_g, ln_b, ynbf);
  // 7) out = yn @ W_out
  dim3 g2(ROWS / BM, DMODEL / BN);                      // 64 x 8
  gemm_bf16_wmma<<<g2, 256, 0, stream>>>(ynbf, woutbf, (float*)d_out, (float*)d_out,
                                         ROWS, DMODEL, DINNER, DMODEL);
}
